// FullBiLSTM_11441792877212
// MI455X (gfx1250) — matrix operand order repaired and checked by reading, not compiled, not run on hardware
//
#include <hip/hip_runtime.h>
#include <cstdint>

#define B_    32
#define T_    512
#define NI_   16384
#define FEAT_ 2048
#define D_    512
#define H_    512
#define G4_   2048   // 4*H
#define NWG_  8      // workgroups per scan cluster (each owns 64 h-columns)

typedef unsigned short u16;
typedef __attribute__((ext_vector_type(16))) __bf16 v16bf;
typedef __attribute__((ext_vector_type(8)))  float  v8f;

union Frag { v16bf v; uint4 q[2]; };

__device__ __forceinline__ u16 f32_to_bf16(float f) {
  union { float f; unsigned u; } x; x.f = f;
  unsigned r = x.u + 0x7fffu + ((x.u >> 16) & 1u);  // round-to-nearest-even
  return (u16)(r >> 16);
}

__device__ __forceinline__ float sigmoidf_(float x) { return 1.0f / (1.0f + __expf(-x)); }
__device__ __forceinline__ float tanhf_(float x)    { return 2.0f * sigmoidf_(2.0f * x) - 1.0f; }

// ---- CDNA5 async copy global->LDS (ASYNCcnt-tracked), per CDNA5 ISA
__device__ __forceinline__ void async_load_b128(unsigned lds_off, const void* gaddr) {
  asm volatile("global_load_async_to_lds_b128 %0, %1, off"
               :: "v"(lds_off), "v"((unsigned long long)(uintptr_t)gaddr) : "memory");
}
__device__ __forceinline__ void wait_async_le4() { asm volatile("s_wait_asynccnt 0x4" ::: "memory"); }
__device__ __forceinline__ void wait_async_0()  { asm volatile("s_wait_asynccnt 0x0" ::: "memory"); }

// A-operand 16x32 bf16 (M x K), row-major K-contiguous (global or LDS).
// lane<16 row=lane, elems0-15=K[k0..k0+15];
// lane>=16 row=lane-16, elems0-15=K[k0+16..k0+31]
__device__ __forceinline__ void load_fragA(Frag& f, const u16* base, int ld, int r0, int k0, int lane) {
  int r  = r0 + (lane & 15);
  int kb = k0 + ((lane >> 4) << 4);
  const uint4* p = (const uint4*)(base + (size_t)r * ld + kb);
  f.q[0] = p[0];
  f.q[1] = p[1];   // +8 bf16 elements
}

// B-operand 32x16 bf16 (K x N) stored as W[N,K] row-major (NT GEMM).
// lane<16: col=lane, K[k0..k0+15]; lane>=16: col=lane-16, K[k0+16..k0+31]
__device__ __forceinline__ void load_fragB(Frag& f, const u16* base, int ld, int n0, int k0, int lane) {
  int n  = n0 + (lane & 15);
  int kb = k0 + ((lane >> 4) << 4);
  const uint4* p = (const uint4*)(base + (size_t)n * ld + kb);
  f.q[0] = p[0];
  f.q[1] = p[1];
}

// ---------------------------------------------------------------- casts
__global__ void cast_f32_bf16(const float* __restrict__ in, u16* __restrict__ out, size_t n) {
  size_t i = (size_t)blockIdx.x * blockDim.x + threadIdx.x;
  if (i < n) out[i] = f32_to_bf16(in[i]);
}

// ------------------------------------------------- NT GEMM, LDS-staged via async copy
// C[M,N] = A[M,K] * B[N,K]^T + bias0 + bias1 ; A,B bf16 ; C fp32
// block = 128 threads = 4 waves; block tile 64x64; K-step 32; double-buffered LDS stage.
__global__ void gemm_nt_bf16(const u16* __restrict__ A, const u16* __restrict__ Bm,
                             const float* __restrict__ bias0, const float* __restrict__ bias1,
                             float* __restrict__ C, int M, int N, int K) {
  __shared__ u16 ldsA[2][64 * 32];
  __shared__ u16 ldsB[2][64 * 32];
  const int tid  = threadIdx.x;
  const int lane = tid & 31;
  const int wave = tid >> 5;
  const int m0b  = blockIdx.x * 64;
  const int n0b  = blockIdx.y * 64;

  // Each 64x32 bf16 tile = 256 x 16B chunks; 128 threads -> 2 chunks each (A and B).
  auto stage = [&](int k, int buf) {
    #pragma unroll
    for (int q = 0; q < 2; ++q) {
      int c = tid * 2 + q;
      int row = c >> 2, piece = c & 3;        // row 0..63, 16B piece 0..3
      const u16* ga = A  + (size_t)(m0b + row) * K + k + piece * 8;
      const u16* gb = Bm + (size_t)(n0b + row) * K + k + piece * 8;
      async_load_b128((unsigned)(uintptr_t)&ldsA[buf][row * 32 + piece * 8], ga);
      async_load_b128((unsigned)(uintptr_t)&ldsB[buf][row * 32 + piece * 8], gb);
    }
  };

  v8f z = {0.f, 0.f, 0.f, 0.f, 0.f, 0.f, 0.f, 0.f};
  v8f acc[4] = {z, z, z, z};
  const int nk = K / 32;
  stage(0, 0);
  for (int ks = 0; ks < nk; ++ks) {
    if (ks + 1 < nk) { stage((ks + 1) * 32, (ks + 1) & 1); wait_async_le4(); }
    else             { wait_async_0(); }
    __syncthreads();                       // stage ks resident for all waves
    const u16* tA = ldsA[ks & 1];
    const u16* tB = ldsB[ks & 1];
    Frag a; load_fragA(a, tA, 32, wave * 16, 0, lane);   // ds_load_b128
    #pragma unroll
    for (int j = 0; j < 4; ++j) {
      Frag b; load_fragB(b, tB, 32, j * 16, 0, lane);
      acc[j] = __builtin_amdgcn_wmma_f32_16x16x32_bf16(false, a.v, false, b.v,
                                                       (short)0, acc[j], false, false);
    }
    __syncthreads();                       // reads done before buffer is re-staged
  }
  const int cc = lane & 15;
  const int cr = (lane >> 4) << 3;
  #pragma unroll
  for (int j = 0; j < 4; ++j) {
    int col = n0b + j * 16 + cc;
    float bs = (bias0 ? bias0[col] : 0.f) + (bias1 ? bias1[col] : 0.f);
    #pragma unroll
    for (int v = 0; v < 8; ++v) {
      int row = m0b + wave * 16 + cr + v;
      C[(size_t)row * N + col] = acc[j][v] + bs;
    }
  }
}

// ------------------------------------------------- stable descending rank sort (B=32)
__global__ void sort_order(const int* __restrict__ lens, int* __restrict__ order,
                           int* __restrict__ lens_s, int* __restrict__ lens_out) {
  int i = threadIdx.x;
  if (i < B_) {
    int li = lens[i];
    int rank = 0;
    for (int j = 0; j < B_; ++j) {
      int lj = lens[j];
      rank += (lj > li) || (lj == li && j < i);   // stable argsort(-lens)
    }
    order[rank] = i;
    lens_s[rank] = li;
    lens_out[rank] = li;
  }
}

// ------------------------------------------------- gather + mask + reverse
__global__ void build_x(const float* __restrict__ feats, const int* __restrict__ lut,
                        const int* __restrict__ order, const int* __restrict__ lens,
                        float* __restrict__ x_out, u16* __restrict__ x_bf,
                        u16* __restrict__ xr_bf) {
  int t = blockIdx.x, b = blockIdx.y;
  int ob = order[b];
  int len = lens[b];
  bool m = (t < len);
  int img  = lut[ob * T_ + t];
  int rt   = m ? (len - 1 - t) : t;
  int rimg = lut[ob * T_ + rt];
  const float* src  = feats + (size_t)img  * D_;
  const float* rsrc = feats + (size_t)rimg * D_;
  size_t base = ((size_t)t * B_ + b) * D_;
  for (int d = threadIdx.x; d < D_; d += blockDim.x) {
    float v  = m ? src[d]  : 0.f;
    float rv = m ? rsrc[d] : 0.f;
    x_out[base + d] = v;
    x_bf[base + d]  = f32_to_bf16(v);
    xr_bf[base + d] = f32_to_bf16(rv);
  }
}

// ------------------------------------------------- cluster-parallel LSTM scan
// NWG_=8 workgroups form one cluster per direction. WG g owns h-columns [g*64,(g+1)*64).
// Per step: gates slice [32,256] = h[32,512] @ Whh[cols]^T + xproj ; pointwise ; write
// h-slice (bf16) to a double-buffered global exchange buffer; s_cluster_barrier; reload
// full h into LDS. (Needs cluster dispatch at runtime; compile-only here.)
__global__ void lstm_scan(const float* __restrict__ xproj, const u16* __restrict__ Whh,
                          const float* __restrict__ h0, const float* __restrict__ c0,
                          const int* __restrict__ lens,
                          float* __restrict__ out, int ostride,
                          float* __restrict__ hn, float* __restrict__ cn,
                          u16* __restrict__ hx /* [2][32*512] bf16 */) {
  extern __shared__ char smem[];
  u16*   hbf   = (u16*)smem;                                  // 32 x 512 bf16 (32 KB)
  float* gates = (float*)(smem + (size_t)B_ * H_ * 2);        // 32 x 256 f32  (32 KB)
  int*   lsh   = (int*)(smem + (size_t)B_ * H_ * 2 + (size_t)B_ * 256 * 4);

  const int tid  = threadIdx.x;          // 256 threads = 8 waves
  const int lane = tid & 31;
  const int wave = tid >> 5;
  const int g    = blockIdx.x;           // workgroup-in-cluster
  const int cc   = lane & 15;
  const int cr   = (lane >> 4) << 3;
  const int jloc = tid & 63;             // local h-column
  const int bs   = tid >> 6;             // batch sub-set 0..3

  if (tid < B_) lsh[tid] = lens[tid];
  for (int q = tid; q < B_ * H_; q += 256) hbf[q] = f32_to_bf16(h0[q]);
  float hreg[8], creg[8];
  #pragma unroll
  for (int r = 0; r < 8; ++r) {
    int b = bs + r * 4;
    int j = g * 64 + jloc;
    hreg[r] = h0[b * H_ + j];
    creg[r] = c0[b * H_ + j];
  }
  __syncthreads();

  for (int t = 0; t < T_; ++t) {
    const float* xp = xproj + (size_t)t * B_ * G4_;
    #pragma unroll
    for (int nti = 0; nti < 2; ++nti) {            // 8 waves x 2 = 16 n-tiles
      const int nt = wave * 2 + nti;
      const int q  = nt >> 2, i = nt & 3;          // gate block q, tile i
      const int n0 = q * 512 + g * 64 + i * 16;    // global gate column
      const int lc = q * 64 + i * 16;              // local column in gates LDS
      v8f a0, a1;
      #pragma unroll
      for (int v = 0; v < 8; ++v) {                // acc = xproj (biases folded in)
        a0[v] = xp[(size_t)(cr + v) * G4_ + n0 + cc];
        a1[v] = xp[(size_t)(16 + cr + v) * G4_ + n0 + cc];
      }
      #pragma unroll
      for (int kk = 0; kk < 16; ++kk) {
        const int k = kk * 32;
        Frag fa0, fa1, fb;
        load_fragA(fa0, hbf, H_, 0, k, lane);      // ds_load_b128 from LDS
        load_fragA(fa1, hbf, H_, 16, k, lane);
        load_fragB(fb, Whh, H_, n0, k, lane);      // streamed from L2
        __builtin_prefetch(Whh + (size_t)(n0 + (lane & 15)) * H_ + k + 64, 0, 1);
        a0 = __builtin_amdgcn_wmma_f32_16x16x32_bf16(false, fa0.v, false, fb.v,
                                                     (short)0, a0, false, false);
        a1 = __builtin_amdgcn_wmma_f32_16x16x32_bf16(false, fa1.v, false, fb.v,
                                                     (short)0, a1, false, false);
      }
      #pragma unroll
      for (int v = 0; v < 8; ++v) {
        gates[(cr + v) * 256 + lc + cc]      = a0[v];
        gates[(16 + cr + v) * 256 + lc + cc] = a1[v];
      }
    }
    __syncthreads();

    u16* hxn = hx + (size_t)((t + 1) & 1) * (B_ * H_);
    #pragma unroll
    for (int r = 0; r < 8; ++r) {
      int b = bs + r * 4;
      float ig = gates[b * 256 + jloc];
      float fg = gates[b * 256 + 64 + jloc];
      float gg = gates[b * 256 + 128 + jloc];
      float og = gates[b * 256 + 192 + jloc];
      ig = sigmoidf_(ig); fg = sigmoidf_(fg); og = sigmoidf_(og); gg = tanhf_(gg);
      float cnew = fg * creg[r] + ig * gg;
      float hnew = og * tanhf_(cnew);
      bool m = (t < lsh[b]);
      out[(size_t)t * B_ * ostride + (size_t)b * ostride + (g * 64 + jloc)] = m ? hnew : 0.f;
      float h2 = m ? hnew : hreg[r];
      float c2 = m ? cnew : creg[r];
      hreg[r] = h2; creg[r] = c2;
      hxn[b * H_ + g * 64 + jloc] = f32_to_bf16(h2);
    }
    __threadfence();                         // make h-slice visible device-scope
    __syncthreads();
    __builtin_amdgcn_s_cluster_barrier();    // s_barrier_signal/-wait -3
    {                                        // reload full h for next step
      const uint4* s = (const uint4*)hxn;
      uint4* d = (uint4*)hbf;
      #pragma unroll
      for (int q2 = 0; q2 < 8; ++q2) d[tid + q2 * 256] = s[tid + q2 * 256];
    }
    __syncthreads();
  }
  #pragma unroll
  for (int r = 0; r < 8; ++r) {
    int b = bs + r * 4;
    int j = g * 64 + jloc;
    hn[b * H_ + j] = hreg[r];
    cn[b * H_ + j] = creg[r];
  }
}

// ------------------------------------------------- un-reverse backward outputs
__global__ void unreverse(const float* __restrict__ outb, const int* __restrict__ lens,
                          float* __restrict__ out) {
  int t = blockIdx.x, b = blockIdx.y;
  int len = lens[b];
  int rt = (t < len) ? (len - 1 - t) : t;
  const float* src = outb + ((size_t)rt * B_ + b) * H_;
  float* dst = out + ((size_t)t * B_ + b) * (2 * H_) + H_;
  for (int j = threadIdx.x; j < H_; j += blockDim.x) dst[j] = src[j];
}

extern "C" void kernel_launch(void* const* d_in, const int* in_sizes, int n_in,
                              void* d_out, int out_size, void* d_ws, size_t ws_size,
                              hipStream_t stream) {
  const float* cnn  = (const float*)d_in[0];
  const float* fcw  = (const float*)d_in[1];
  const float* fcb  = (const float*)d_in[2];
  const float* h0   = (const float*)d_in[3];
  const float* c0   = (const float*)d_in[4];
  const float* wihf = (const float*)d_in[5];
  const float* whhf = (const float*)d_in[6];
  const float* bihf = (const float*)d_in[7];
  const float* bhhf = (const float*)d_in[8];
  const float* wihb = (const float*)d_in[9];
  const float* whhb = (const float*)d_in[10];
  const float* bihb = (const float*)d_in[11];
  const float* bhhb = (const float*)d_in[12];
  const int*   slens = (const int*)d_in[13];
  const int*   lut   = (const int*)d_in[14];

  char* w = (char*)d_ws;
  u16* cnn_bf  = (u16*)w;   w += (size_t)NI_ * FEAT_ * 2;
  u16* fcw_bf  = (u16*)w;   w += (size_t)D_ * FEAT_ * 2;
  u16* wihf_bf = (u16*)w;   w += (size_t)G4_ * D_ * 2;
  u16* whhf_bf = (u16*)w;   w += (size_t)G4_ * H_ * 2;
  u16* wihb_bf = (u16*)w;   w += (size_t)G4_ * D_ * 2;
  u16* whhb_bf = (u16*)w;   w += (size_t)G4_ * H_ * 2;
  float* feats = (float*)w; w += (size_t)NI_ * D_ * 4;
  u16* x_bf    = (u16*)w;   w += (size_t)T_ * B_ * D_ * 2;
  u16* xr_bf   = (u16*)w;   w += (size_t)T_ * B_ * D_ * 2;
  float* xpf   = (float*)w; w += (size_t)T_ * B_ * G4_ * 4;
  float* xpb   = (float*)w; w += (size_t)T_ * B_ * G4_ * 4;
  float* obrev = (float*)w; w += (size_t)T_ * B_ * H_ * 4;
  int* order   = (int*)w;   w += 128;
  int* lens_s  = (int*)w;   w += 128;
  u16* hxf     = (u16*)w;   w += (size_t)2 * B_ * H_ * 2;  // h exchange, fwd (double buf)
  u16* hxb     = (u16*)w;   w += (size_t)2 * B_ * H_ * 2;  // h exchange, bwd

  float* out_seq = (float*)d_out;                       // [T,B,2H]
  float* hn = out_seq + (size_t)T_ * B_ * 2 * H_;       // [2,B,H]
  float* cn = hn + (size_t)2 * B_ * H_;                 // [2,B,H]
  float* x_out = cn + (size_t)2 * B_ * H_;              // [T,B,D]
  int* lens_out = (int*)(x_out + (size_t)T_ * B_ * D_); // [B] int32

  // fp32 -> bf16 conversions
  cast_f32_bf16<<<dim3((NI_ * FEAT_ + 511) / 512), 512, 0, stream>>>(cnn, cnn_bf, (size_t)NI_ * FEAT_);
  cast_f32_bf16<<<dim3((D_ * FEAT_ + 511) / 512), 512, 0, stream>>>(fcw, fcw_bf, (size_t)D_ * FEAT_);
  cast_f32_bf16<<<dim3((G4_ * D_ + 511) / 512), 512, 0, stream>>>(wihf, wihf_bf, (size_t)G4_ * D_);
  cast_f32_bf16<<<dim3((G4_ * H_ + 511) / 512), 512, 0, stream>>>(whhf, whhf_bf, (size_t)G4_ * H_);
  cast_f32_bf16<<<dim3((G4_ * D_ + 511) / 512), 512, 0, stream>>>(wihb, wihb_bf, (size_t)G4_ * D_);
  cast_f32_bf16<<<dim3((G4_ * H_ + 511) / 512), 512, 0, stream>>>(whhb, whhb_bf, (size_t)G4_ * H_);

  // feats = cnn_feats @ fc_w^T + fc_b
  gemm_nt_bf16<<<dim3(NI_ / 64, D_ / 64), 128, 0, stream>>>(cnn_bf, fcw_bf, fcb, nullptr,
                                                            feats, NI_, D_, FEAT_);
  // sort order + lens
  sort_order<<<1, 32, 0, stream>>>(slens, order, lens_s, lens_out);

  // gather + mask + reverse
  build_x<<<dim3(T_, B_), 256, 0, stream>>>(feats, lut, order, lens_s, x_out, x_bf, xr_bf);

  // precompute input projections (biases folded in)
  gemm_nt_bf16<<<dim3(T_ * B_ / 64, G4_ / 64), 128, 0, stream>>>(x_bf, wihf_bf, bihf, bhhf,
                                                                 xpf, T_ * B_, G4_, D_);
  gemm_nt_bf16<<<dim3(T_ * B_ / 64, G4_ / 64), 128, 0, stream>>>(xr_bf, wihb_bf, bihb, bhhb,
                                                                 xpb, T_ * B_, G4_, D_);

  // recurrent scans: one 8-WG cluster per direction; 64 KB LDS per WG
  size_t smem = (size_t)B_ * H_ * 2 + (size_t)B_ * 256 * 4 + 128;
  lstm_scan<<<NWG_, 256, smem, stream>>>(xpf, whhf_bf, h0, c0, lens_s,
                                         out_seq, 2 * H_, hn, cn, hxf);
  lstm_scan<<<NWG_, 256, smem, stream>>>(xpb, whhb_bf, h0 + B_ * H_, c0 + B_ * H_, lens_s,
                                         obrev, H_, hn + B_ * H_, cn + B_ * H_, hxb);

  // scatter backward half into out[:, :, H:2H]
  unreverse<<<dim3(T_, B_), 256, 0, stream>>>(obrev, lens_s, out_seq);
}
